// SS3D_v6_50371376447796
// MI455X (gfx1250) — compile-verified, matrix operand order checked by reading, not run on hardware
//
#include <hip/hip_runtime.h>
#include <hip/hip_bf16.h>

typedef __attribute__((ext_vector_type(16))) _Float16 v16h;
typedef __attribute__((ext_vector_type(8)))  float    v8f;

#define L_SEQ 4096
#define CI_CH 96
#define ED_CH 192
#define NS    16
#define NCHUNK 32
#define CLEN  128   // L_SEQ / NCHUNK

// padded dbc layout: cols 0..5 = dt, 8..23 = B, 24..39 = C, rest zero (ld = 48)
#define DBC_LD 48
#define DBC_B  8
#define DBC_C  24

// ---------------------------------------------------------------------------
// helpers
// ---------------------------------------------------------------------------
__device__ __forceinline__ float siluf(float x) {
    return x / (1.0f + __expf(-x));
}

// sequence-order -> spatial (h,w,d) flat index mapping for the 6 scan branches
// order/2: 0 = (H,W,D), 1 = (D,W,H), 2 = (H,D,W);   order&1: reversed sequence
__device__ __forceinline__ int spatial_index(int order, int pos) {
    int p = (order & 1) ? (L_SEQ - 1 - pos) : pos;
    int a = (p >> 8) & 15, b = (p >> 4) & 15, c = p & 15;
    int h, w, d;
    int base = order >> 1;
    if (base == 0)      { h = a; w = b; d = c; }
    else if (base == 1) { d = a; w = b; h = c; }
    else                { h = a; d = b; w = c; }
    return (h * 16 + w) * 16 + d;
}

// ---------------------------------------------------------------------------
// WMMA fragment loaders (layouts per CDNA5 ISA 7.12.2, f16 16x16x32).
// Per lane the fragment is two contiguous 8-float runs -> four float4 loads.
// A: 16x32 (MxK). lane 0-15: M=lane, halves 0-7 -> K 0-7, halves 8-15 -> K 16-23
//                 lane 16-31: same M, K ranges +8.
// ---------------------------------------------------------------------------
__device__ __forceinline__ v16h load_frag_rowmajor(const float* __restrict__ P,
                                                   int ld, int r0, int k0, int lane) {
    int r = r0 + (lane & 15);
    int kbase = k0 + ((lane >> 4) << 3);
    const float* p = P + (size_t)r * ld + kbase;
    float4 q0 = *(const float4*)(p);
    float4 q1 = *(const float4*)(p + 4);
    float4 q2 = *(const float4*)(p + 16);
    float4 q3 = *(const float4*)(p + 20);
    v16h f;
    f[0]  = (_Float16)q0.x; f[1]  = (_Float16)q0.y;
    f[2]  = (_Float16)q0.z; f[3]  = (_Float16)q0.w;
    f[4]  = (_Float16)q1.x; f[5]  = (_Float16)q1.y;
    f[6]  = (_Float16)q1.z; f[7]  = (_Float16)q1.w;
    f[8]  = (_Float16)q2.x; f[9]  = (_Float16)q2.y;
    f[10] = (_Float16)q2.z; f[11] = (_Float16)q2.w;
    f[12] = (_Float16)q3.x; f[13] = (_Float16)q3.y;
    f[14] = (_Float16)q3.z; f[15] = (_Float16)q3.w;
    return f;
}

// ---------------------------------------------------------------------------
// Generic WMMA GEMM: C[4096 x N] = A[4096 x K] * W^T   (W row-major (N,K))
// 8 waves / block, one 16x16 tile per wave; exact grids -> EXEC all-ones.
// ---------------------------------------------------------------------------
template <int K>
__global__ void wmma_gemm_wt(const float* __restrict__ A,
                             const float* __restrict__ W,
                             float* __restrict__ C,
                             int Ntiles, int lda, int ldw, int ldc) {
    int wave = threadIdx.x >> 5;
    int lane = threadIdx.x & 31;
    int tile = blockIdx.x * 8 + wave;
    int m0 = (tile / Ntiles) * 16;
    int n0 = (tile % Ntiles) * 16;

    v8f acc = {};
#pragma unroll
    for (int k0 = 0; k0 < K; k0 += 32) {
        v16h a = load_frag_rowmajor(A, lda, m0, k0, lane);
        v16h b = load_frag_rowmajor(W, ldw, n0, k0, lane);
        acc = __builtin_amdgcn_wmma_f32_16x16x32_f16(
            false, a, false, b, (short)0, acc, false, false);
    }

    // D layout: lane&15 -> N, vgpr r -> M = r (+8 for lanes 16-31)
    int n = n0 + (lane & 15);
    int mb = m0 + ((lane >> 4) << 3);
#pragma unroll
    for (int r = 0; r < 8; ++r) {
        C[(size_t)(mb + r) * ldc + n] = acc[r];
    }
}

// m_out GEMM fused with residual add + permuted scatter-accumulate into acc.
// out_s[pos][c] = uS[pos][c] + (ym @ m_out_w^T)[pos][c];  acc[spatial] += out_s
__global__ void wmma_gemm_out_scatter(const float* __restrict__ ym,
                                      const float* __restrict__ W,
                                      const float* __restrict__ uS,
                                      float* __restrict__ acc, int order) {
    int wave = threadIdx.x >> 5;
    int lane = threadIdx.x & 31;
    int tile = blockIdx.x * 8 + wave;
    const int Ntiles = CI_CH / 16;   // 6
    int m0 = (tile / Ntiles) * 16;
    int n0 = (tile % Ntiles) * 16;

    v8f accf = {};
#pragma unroll
    for (int k0 = 0; k0 < ED_CH; k0 += 32) {
        v16h a = load_frag_rowmajor(ym, ED_CH, m0, k0, lane);
        v16h b = load_frag_rowmajor(W, ED_CH, n0, k0, lane);
        accf = __builtin_amdgcn_wmma_f32_16x16x32_f16(
            false, a, false, b, (short)0, accf, false, false);
    }

    int n = n0 + (lane & 15);
    int mb = m0 + ((lane >> 4) << 3);
#pragma unroll
    for (int r = 0; r < 8; ++r) {
        int m = mb + r;
        int sp = spatial_index(order, m);
        acc[(size_t)sp * CI_CH + n] += uS[(size_t)m * CI_CH + n] + accf[r];
    }
}

// ---------------------------------------------------------------------------
// Pad + permute x_proj_w (38 x 192) into 48 x 192 so dbc cols are:
// 0..5 = dt rows, 8..23 = B rows, 24..39 = C rows, rest zero.
// ---------------------------------------------------------------------------
__global__ void pad_xproj(const float* __restrict__ W, float* __restrict__ Wp) {
    int tid = blockIdx.x * blockDim.x + threadIdx.x;   // exact: 48*192
    int r = tid / ED_CH;
    int k = tid % ED_CH;
    int orig = (r < 6) ? r : ((r >= DBC_B && r < DBC_C + NS) ? (r - 2) : -1);
    Wp[tid] = (orig >= 0) ? W[(size_t)orig * ED_CH + k] : 0.0f;
}

// ---------------------------------------------------------------------------
// Depthwise 3x3x3 conv over (H,W,D) + bias + SiLU.  h, x0: [spatial][96]
// ---------------------------------------------------------------------------
__global__ void dwconv3d_silu(const float* __restrict__ h,
                              const float* __restrict__ w3,
                              const float* __restrict__ b3,
                              float* __restrict__ x0) {
    int tid = blockIdx.x * blockDim.x + threadIdx.x;   // exact: 4096*96
    int c  = tid % CI_CH;
    int sp = tid / CI_CH;
    int d  = sp & 15, w = (sp >> 4) & 15, hh = (sp >> 8) & 15;
    float s = b3[c];
    const float* wk = w3 + c * 27;
#pragma unroll
    for (int kh = 0; kh < 3; ++kh) {
        int ih = hh + kh - 1;
        if ((unsigned)ih >= 16u) continue;
#pragma unroll
        for (int kw = 0; kw < 3; ++kw) {
            int iw = w + kw - 1;
            if ((unsigned)iw >= 16u) continue;
#pragma unroll
            for (int kd = 0; kd < 3; ++kd) {
                int id = d + kd - 1;
                if ((unsigned)id >= 16u) continue;
                s += wk[kh * 9 + kw * 3 + kd] *
                     h[(size_t)((ih * 16 + iw) * 16 + id) * CI_CH + c];
            }
        }
    }
    x0[tid] = siluf(s);
}

// ---------------------------------------------------------------------------
// RMS-norm + gather into sequence order.  One wave per token (8 waves/block).
// ---------------------------------------------------------------------------
__global__ void rms_gather(const float* __restrict__ x0,
                           const float* __restrict__ rms_w,
                           float* __restrict__ uS, float* __restrict__ xn,
                           int order) {
    int wave = threadIdx.x >> 5, lane = threadIdx.x & 31;
    int pos = blockIdx.x * 8 + wave;
    int sp = spatial_index(order, pos);
    float u[3];
    float ss = 0.0f;
#pragma unroll
    for (int i = 0; i < 3; ++i) {
        int c = lane + i * 32;
        u[i] = x0[(size_t)sp * CI_CH + c];
        ss += u[i] * u[i];
    }
#pragma unroll
    for (int m = 16; m > 0; m >>= 1) ss += __shfl_xor(ss, m, 32);
    float inv = rsqrtf(ss * (1.0f / CI_CH) + 1e-5f);
#pragma unroll
    for (int i = 0; i < 3; ++i) {
        int c = lane + i * 32;
        uS[(size_t)pos * CI_CH + c] = u[i];
        xn[(size_t)pos * CI_CH + c] = u[i] * inv * rms_w[c];
    }
}

// ---------------------------------------------------------------------------
// Causal depthwise conv1d (width 4) + SiLU.  xz:[t][384] (xm = cols 0..191)
// ---------------------------------------------------------------------------
__global__ void conv1d_silu(const float* __restrict__ xz,
                            const float* __restrict__ cw,
                            const float* __restrict__ cb,
                            float* __restrict__ xma) {
    int tid = blockIdx.x * blockDim.x + threadIdx.x;   // exact: 4096*192
    int e = tid % ED_CH;
    int t = tid / ED_CH;
    float s = cb[e];
#pragma unroll
    for (int j = 0; j < 4; ++j) {
        int ts = t - 3 + j;
        if (ts >= 0) s += cw[e * 4 + j] * xz[(size_t)ts * 384 + e];
    }
    xma[tid] = siluf(s);
}

// ---------------------------------------------------------------------------
// delta = softplus(dt @ dt_proj_w^T + b)   (K = DT_RANK = 6, scalar path)
// ---------------------------------------------------------------------------
__global__ void dt_softplus(const float* __restrict__ dbc,
                            const float* __restrict__ dtw,
                            const float* __restrict__ dtb,
                            float* __restrict__ delta) {
    int tid = blockIdx.x * blockDim.x + threadIdx.x;   // exact: 4096*192
    int e = tid % ED_CH;
    int t = tid / ED_CH;
    float s = dtb[e];
#pragma unroll
    for (int r = 0; r < 6; ++r) s += dtw[e * 6 + r] * dbc[(size_t)t * DBC_LD + r];
    delta[tid] = (s > 20.0f) ? s : __logf(1.0f + __expf(s));
}

// aligned float4 row fetch of 16 consecutive floats into a register array
__device__ __forceinline__ void load16(const float* __restrict__ p, float* v) {
    float4 a = *(const float4*)(p);
    float4 b = *(const float4*)(p + 4);
    float4 c = *(const float4*)(p + 8);
    float4 d = *(const float4*)(p + 12);
    v[0]=a.x; v[1]=a.y; v[2]=a.z; v[3]=a.w;
    v[4]=b.x; v[5]=b.y; v[6]=b.z; v[7]=b.w;
    v[8]=c.x; v[9]=c.y; v[10]=c.z; v[11]=c.w;
    v[12]=d.x; v[13]=d.y; v[14]=d.z; v[15]=d.w;
}

// ---------------------------------------------------------------------------
// Chunked selective scan, phase 1: per (e, chunk) compute chunk-local
// (prod dA, partial state) for all 16 states.
// ---------------------------------------------------------------------------
__global__ void scan_phase1(const float* __restrict__ delta,
                            const float* __restrict__ xma,
                            const float* __restrict__ dbc,
                            const float* __restrict__ A_log,
                            float* __restrict__ chA, float* __restrict__ chB) {
    int tid = blockIdx.x * blockDim.x + threadIdx.x;   // exact: 192*32
    int e = tid % ED_CH;
    int chunk = tid / ED_CH;
    float Ae[NS], ap[NS], bb[NS];
#pragma unroll
    for (int n = 0; n < NS; ++n) {
        Ae[n] = -__expf(A_log[e * NS + n]);
        ap[n] = 1.0f;
        bb[n] = 0.0f;
    }
    int t0 = chunk * CLEN;
    for (int tt = 0; tt < CLEN; ++tt) {
        int t = t0 + tt;
        float dt = delta[(size_t)t * ED_CH + e];
        float dx = dt * xma[(size_t)t * ED_CH + e];
        float bv[NS];
        load16(dbc + (size_t)t * DBC_LD + DBC_B, bv);
#pragma unroll
        for (int n = 0; n < NS; ++n) {
            float a = __expf(dt * Ae[n]);
            ap[n] *= a;
            bb[n] = a * bb[n] + dx * bv[n];
        }
    }
    size_t base = ((size_t)e * NCHUNK + chunk) * NS;
#pragma unroll
    for (int n = 0; n < NS; ++n) { chA[base + n] = ap[n]; chB[base + n] = bb[n]; }
}

// Phase 2: serial 32-step prefix across chunks per (e,n) state.
__global__ void scan_phase2(const float* __restrict__ chA,
                            const float* __restrict__ chB,
                            float* __restrict__ hIn) {
    int tid = blockIdx.x * blockDim.x + threadIdx.x;   // exact: 192*16
    int e = tid >> 4, n = tid & 15;
    float h = 0.0f;
    for (int c = 0; c < NCHUNK; ++c) {
        size_t idx = ((size_t)e * NCHUNK + c) * NS + n;
        hIn[idx] = h;
        h = chA[idx] * h + chB[idx];
    }
}

// Phase 3: re-scan each chunk with its prefix state; emit gated y.
__global__ void scan_phase3(const float* __restrict__ delta,
                            const float* __restrict__ xma,
                            const float* __restrict__ dbc,
                            const float* __restrict__ xz,
                            const float* __restrict__ A_log,
                            const float* __restrict__ Dp,
                            const float* __restrict__ hIn,
                            float* __restrict__ ym) {
    int tid = blockIdx.x * blockDim.x + threadIdx.x;   // exact: 192*32
    int e = tid % ED_CH;
    int chunk = tid / ED_CH;
    float Ae[NS], h[NS];
    size_t base = ((size_t)e * NCHUNK + chunk) * NS;
#pragma unroll
    for (int n = 0; n < NS; ++n) {
        Ae[n] = -__expf(A_log[e * NS + n]);
        h[n] = hIn[base + n];
    }
    float De = Dp[e];
    int t0 = chunk * CLEN;
    for (int tt = 0; tt < CLEN; ++tt) {
        int t = t0 + tt;
        float dt = delta[(size_t)t * ED_CH + e];
        float xv = xma[(size_t)t * ED_CH + e];
        float dx = dt * xv;
        float bv[NS], cv[NS];
        load16(dbc + (size_t)t * DBC_LD + DBC_B, bv);
        load16(dbc + (size_t)t * DBC_LD + DBC_C, cv);
        float ysum = 0.0f;
#pragma unroll
        for (int n = 0; n < NS; ++n) {
            float a = __expf(dt * Ae[n]);
            h[n] = a * h[n] + dx * bv[n];
            ysum += h[n] * cv[n];
        }
        float y = ysum + De * xv;
        float zv = xz[(size_t)t * 384 + ED_CH + e];
        ym[(size_t)t * ED_CH + e] = y * siluf(zv);
    }
}

// ---------------------------------------------------------------------------
// LayerNorm over 96 channels. One wave per token.
// ---------------------------------------------------------------------------
__global__ void layernorm96(const float* __restrict__ acc,
                            const float* __restrict__ g,
                            const float* __restrict__ b,
                            float* __restrict__ out) {
    int wave = threadIdx.x >> 5, lane = threadIdx.x & 31;
    int pos = blockIdx.x * 8 + wave;
    float v[3];
    float s1 = 0.0f, s2 = 0.0f;
#pragma unroll
    for (int i = 0; i < 3; ++i) {
        int c = lane + i * 32;
        v[i] = acc[(size_t)pos * CI_CH + c];
        s1 += v[i];
        s2 += v[i] * v[i];
    }
#pragma unroll
    for (int m = 16; m > 0; m >>= 1) {
        s1 += __shfl_xor(s1, m, 32);
        s2 += __shfl_xor(s2, m, 32);
    }
    float mu = s1 * (1.0f / CI_CH);
    float var = s2 * (1.0f / CI_CH) - mu * mu;
    float inv = rsqrtf(var + 1e-5f);
#pragma unroll
    for (int i = 0; i < 3; ++i) {
        int c = lane + i * 32;
        out[(size_t)pos * CI_CH + c] = (v[i] - mu) * inv * g[c] + b[c];
    }
}

// ---------------------------------------------------------------------------
// launch
// ---------------------------------------------------------------------------
extern "C" void kernel_launch(void* const* d_in, const int* in_sizes, int n_in,
                              void* d_out, int out_size, void* d_ws, size_t ws_size,
                              hipStream_t stream) {
    (void)in_sizes; (void)n_in; (void)out_size; (void)ws_size;
    const float* x         = (const float*)d_in[0];
    const float* in_proj_w = (const float*)d_in[1];
    const float* conv3d_w  = (const float*)d_in[2];
    const float* conv3d_b  = (const float*)d_in[3];
    const float* rms_w     = (const float*)d_in[4];
    const float* m_in_w    = (const float*)d_in[5];
    const float* m_conv_w  = (const float*)d_in[6];
    const float* m_conv_b  = (const float*)d_in[7];
    const float* x_proj_w  = (const float*)d_in[8];
    const float* dt_proj_w = (const float*)d_in[9];
    const float* dt_proj_b = (const float*)d_in[10];
    const float* A_log     = (const float*)d_in[11];
    const float* D_param   = (const float*)d_in[12];
    const float* m_out_w   = (const float*)d_in[13];
    const float* ln_g      = (const float*)d_in[14];
    const float* ln_b      = (const float*)d_in[15];
    const float* out_proj_w= (const float*)d_in[16];
    float* out = (float*)d_out;

    float* ws = (float*)d_ws;
    size_t o = 0;
    float* hbuf  = ws + o; o += (size_t)L_SEQ * CI_CH;       // reused as LN output
    float* x0b   = ws + o; o += (size_t)L_SEQ * CI_CH;
    float* accb  = ws + o; o += (size_t)L_SEQ * CI_CH;
    float* uS    = ws + o; o += (size_t)L_SEQ * CI_CH;
    float* xn    = ws + o; o += (size_t)L_SEQ * CI_CH;
    float* xz    = ws + o; o += (size_t)L_SEQ * 384;
    float* xma   = ws + o; o += (size_t)L_SEQ * ED_CH;
    float* dbc   = ws + o; o += (size_t)L_SEQ * DBC_LD;
    float* delta = ws + o; o += (size_t)L_SEQ * ED_CH;
    float* ym    = ws + o; o += (size_t)L_SEQ * ED_CH;
    float* chA   = ws + o; o += (size_t)ED_CH * NCHUNK * NS;
    float* chB   = ws + o; o += (size_t)ED_CH * NCHUNK * NS;
    float* hIn   = ws + o; o += (size_t)ED_CH * NCHUNK * NS;
    float* xprojp= ws + o; o += (size_t)48 * ED_CH;          // padded x_proj_w

    const int BLK = 256;

    // pad/permute x_proj weights (48*192 = 9216 -> 36 blocks exact)
    pad_xproj<<<36, BLK, 0, stream>>>(x_proj_w, xprojp);
    // h = x @ in_proj_w^T   (M=4096, K=96, N=96)
    wmma_gemm_wt<96><<<192, BLK, 0, stream>>>(x, in_proj_w, hbuf, 6, 96, 96, 96);
    // depthwise 3x3x3 conv + bias + SiLU -> x0
    dwconv3d_silu<<<(L_SEQ * CI_CH) / BLK, BLK, 0, stream>>>(hbuf, conv3d_w, conv3d_b, x0b);
    // zero branch accumulator
    hipMemsetAsync(accb, 0, (size_t)L_SEQ * CI_CH * sizeof(float), stream);

    for (int order = 0; order < 6; ++order) {
        // RMS norm + gather to sequence order
        rms_gather<<<L_SEQ / 8, BLK, 0, stream>>>(x0b, rms_w, uS, xn, order);
        // xz = xn @ m_in_w^T  (K=96, N=384)
        wmma_gemm_wt<96><<<768, BLK, 0, stream>>>(xn, m_in_w, xz, 24, 96, 96, 384);
        // causal conv1d width 4 + SiLU
        conv1d_silu<<<(L_SEQ * ED_CH) / BLK, BLK, 0, stream>>>(xz, m_conv_w, m_conv_b, xma);
        // dbc = xma @ xprojp^T  (K=192, N=48 padded/permuted)
        wmma_gemm_wt<192><<<96, BLK, 0, stream>>>(xma, xprojp, dbc, 3, 192, 192, DBC_LD);
        // delta = softplus(dt @ dt_proj_w^T + b)
        dt_softplus<<<(L_SEQ * ED_CH) / BLK, BLK, 0, stream>>>(dbc, dt_proj_w, dt_proj_b, delta);
        // chunked selective scan
        scan_phase1<<<(ED_CH * NCHUNK) / BLK, BLK, 0, stream>>>(delta, xma, dbc, A_log, chA, chB);
        scan_phase2<<<(ED_CH * NS) / BLK, BLK, 0, stream>>>(chA, chB, hIn);
        scan_phase3<<<(ED_CH * NCHUNK) / BLK, BLK, 0, stream>>>(delta, xma, dbc, xz, A_log,
                                                               D_param, hIn, ym);
        // out = u + ym @ m_out_w^T, scatter-accumulate into spatial frame
        wmma_gemm_out_scatter<<<192, BLK, 0, stream>>>(ym, m_out_w, uS, accb, order);
    }

    // LayerNorm over channels, then final projection to d_out
    layernorm96<<<L_SEQ / 8, BLK, 0, stream>>>(accb, ln_g, ln_b, hbuf);
    wmma_gemm_wt<96><<<192, BLK, 0, stream>>>(hbuf, out_proj_w, out, 6, 96, 96, 96);
}
